// DecoderRNN_29884382446196
// MI455X (gfx1250) — matrix operand order repaired and checked by reading, not compiled, not run on hardware
//
#include <hip/hip_runtime.h>
#include <hip/hip_bf16.h>

// ---------------------------------------------------------------------------
// Sequential LSTM decode (T=1024, H=2048, batch=1) for MI455X / gfx1250.
//
// Per-step cost = streaming W_hh from L2 (192MB L2 holds it). W_hh is stored
// once per launch as bf16 (32MB/step vs 64MB fp32), pre-swizzled into the
// v_wmma bf16 A-fragment layout so each lane's A load is 32 contiguous bytes.
// GEMV via v_wmma_f32_16x16x32_bf16 with the h-chunk in column 0 of B; the
// unused D columns are never read, so all 32 lanes load h uniformly (no exec
// masking, no zero-fill) -- garbage columns are harmless.
// One persistent kernel, 64 blocks x 1024 threads (32 waves): 4 gates x 2
// row-subtiles x 4 K-quarters per block; K-partials combined through LDS.
// Device-wide monotonic spin barrier per timestep keeps h/c on-chip.
// ---------------------------------------------------------------------------

typedef __attribute__((ext_vector_type(16))) __bf16 v16bf;
typedef __attribute__((ext_vector_type(8)))  float  v8f;

namespace {
constexpr int H     = 2048;
constexpr int G4    = 4 * H;        // 8192 gate rows
constexpr int T_LEN = 1024;
constexpr int NB    = 64;           // persistent blocks (guaranteed resident)
constexpr int TPB   = 1024;         // 32 waves (wave32)
constexpr int KQ    = 4;            // K-quarters per row-tile
constexpr int KC_PER_Q = (H / 32) / KQ;   // 16 WMMAs per wave per step
constexpr int TILE_ELEMS = 16 * 32; // 512 bf16 per A fragment (1KB)

constexpr size_t WSWZ_BYTES = (size_t)G4 * H * 2;            // 32 MB
constexpr size_t HBUF_BYTES = 2 * H * 2;                     // 8 KB
constexpr size_t BAR_OFF    = WSWZ_BYTES + HBUF_BYTES;
constexpr size_t PART_OFF   = BAR_OFF + 1024;
}

// Swizzle W_hh (fp32, row-major [8192][2048]) into bf16 WMMA A-fragments.
// Fragment layout (16-bit A 16x32, contiguous K to match B):
//   lanes 0-15 : VGPR0..7 = K 0..15   (row M = lane)
//   lanes 16-31: VGPR0..7 = K 16..31  (row M = lane-16)
// Stored so that lane l reads 16 contiguous halves at tile*512 + l*16.
__global__ void convert_whh_kernel(const float* __restrict__ Whh,
                                   __bf16* __restrict__ Wswz) {
    size_t e = (size_t)blockIdx.x * blockDim.x + threadIdx.x; // < 8192*2048
    int tile   = (int)(e >> 9);      // / 512
    int within = (int)(e & 511);
    int lane = within >> 4;          // 0..31
    int h8   = within & 15;          // which half within the lane's 16
    int rt = tile >> 6;              // row-tile 0..511
    int kc = tile & 63;              // k-chunk 0..63
    int kl = h8;
    if (lane >= 16) kl += 16;
    int row = rt * 16 + (lane & 15);
    int k   = kc * 32 + kl;
    Wswz[e] = (__bf16)Whh[(size_t)row * H + k];
}

__global__ void init_kernel(const float* __restrict__ feat,
                            __bf16* __restrict__ hbuf,
                            unsigned* __restrict__ bar) {
    int i = blockIdx.x * blockDim.x + threadIdx.x;
    if (i < H) hbuf[i] = (__bf16)feat[i];  // h0 = features (buffer 0)
    if (i == 0) *bar = 0u;
}

__global__ __launch_bounds__(TPB, 1)
void lstm_persistent_kernel(const __bf16* __restrict__ Wswz,
                            __bf16* __restrict__ hbuf,        // [2][H]
                            const float* __restrict__ pc,     // [T][2]
                            const float* __restrict__ W_ih,   // [8192][2]
                            const float* __restrict__ b_ih,   // [8192]
                            const float* __restrict__ b_hh,   // [8192]
                            const float* __restrict__ W_fc,   // [2][H]
                            const float* __restrict__ b_fc,   // [2]
                            float* __restrict__ partials,     // [T][NB][2]
                            float* __restrict__ out,          // [T][2]
                            unsigned* __restrict__ bar) {
    const int b   = blockIdx.x;     // 0..63
    const int tid = threadIdx.x;
    const int w   = tid >> 5;       // wave 0..31
    const int l   = tid & 31;       // lane
    const int gsel = w >> 3;        // gate: 0=i 1=f 2=g 3=o (PyTorch order)
    const int tsel = (w >> 2) & 1;  // row-subtile within the block's 32 rows
    const int kq   = w & 3;         // K-quarter (each covers K=512)
    const int rt = gsel * 128 + b * 2 + tsel;  // row-tile index 0..511
    const int R  = rt * 16;                    // first gate row of this tile

    __shared__ float g_parts[4][KQ][32];       // [gate][k-quarter][row]

    float c_reg = 0.0f;             // cell state, lives in wave-0 lanes forever
    const int j = b * 32 + l;       // owned h index (tid < 32)

    const __bf16* Abase = Wswz + (size_t)rt * (64 * TILE_ELEMS)
                               + (size_t)kq * (KC_PER_Q * TILE_ELEMS)
                               + l * 16;
    const int hq = kq * KC_PER_Q * 32 + (l & 16);  // per-lane h offset base

    for (int t = 0; t < T_LEN; ++t) {
        const __bf16* hprev = hbuf + (size_t)(t & 1) * H;

        // ---- GEMV tile: D[:,0] += W_tile[16 x 512] * h[quarter] ----
        // All 32 lanes load h uniformly; only lanes 0/16 feed column 0 of B,
        // the other columns of D are discarded, so no zeroing/exec-masking.
        v8f acc = {};
#pragma unroll 4
        for (int kc = 0; kc < KC_PER_Q; ++kc) {
            v16bf a  = *(const v16bf*)(Abase + (size_t)kc * TILE_ELEMS);
            v16bf bb = *(const v16bf*)(hprev + hq + kc * 32);
            acc = __builtin_amdgcn_wmma_f32_16x16x32_bf16(
                false, a, false, bb, (short)0, acc, false, false);
        }

        // Column 0 of D sits in lanes 0 (M=0..7) and 16 (M=8..15).
        if (l == 0 || l == 16) {
            const int off = l ? 8 : 0;
            if (kq == 0) {
                const float x0 = pc[t * 2 + 0];
                const float x1 = pc[t * 2 + 1];
#pragma unroll
                for (int i = 0; i < 8; ++i) {
                    const int r = R + off + i;
                    g_parts[gsel][0][tsel * 16 + off + i] =
                        acc[i] + b_ih[r] + b_hh[r]
                        + x0 * W_ih[r * 2] + x1 * W_ih[r * 2 + 1];
                }
            } else {
#pragma unroll
                for (int i = 0; i < 8; ++i)
                    g_parts[gsel][kq][tsel * 16 + off + i] = acc[i];
            }
        }
        __syncthreads();

        // ---- pointwise cell update on wave 0 (one lane per owned h elem) ----
        if (tid < 32) {
            float gv[4];
#pragma unroll
            for (int g = 0; g < 4; ++g) {
                float s = g_parts[g][0][l];
#pragma unroll
                for (int q = 1; q < KQ; ++q) s += g_parts[g][q][l];
                gv[g] = s;
            }
            float si = 1.0f / (1.0f + expf(-gv[0]));
            float sf = 1.0f / (1.0f + expf(-gv[1]));
            float so = 1.0f / (1.0f + expf(-gv[3]));
            float c_new = sf * c_reg + si * tanhf(gv[2]);
            float h_new = so * tanhf(c_new);
            c_reg = c_new;
            hbuf[(size_t)((t + 1) & 1) * H + j] = (__bf16)h_new;

            // fc partials for this block's 32 h elements (fp32, deterministic)
            float p0 = h_new * W_fc[j];
            float p1 = h_new * W_fc[H + j];
#pragma unroll
            for (int m = 16; m > 0; m >>= 1) {
                p0 += __shfl_xor(p0, m, 32);
                p1 += __shfl_xor(p1, m, 32);
            }
            if (l == 0) {
                partials[(size_t)(t * NB + b) * 2 + 0] = p0;
                partials[(size_t)(t * NB + b) * 2 + 1] = p1;
            }
        }
        __syncthreads();

        // ---- device-wide step barrier (monotonic counter, no reset) ----
        if (tid == 0) {
            __threadfence();                         // release h/partials
            atomicAdd(bar, 1u);
            const unsigned target = (unsigned)(NB * (t + 1));
            while (__hip_atomic_load(bar, __ATOMIC_RELAXED,
                                     __HIP_MEMORY_SCOPE_AGENT) < target)
                __builtin_amdgcn_s_sleep(2);
        }
        __syncthreads();
        __threadfence();                             // acquire: invalidate L0
    }

    // ---- final deterministic reduction of fc partials into d_out ----
    int gidx = b * TPB + tid;
    if (gidx < T_LEN * 2) {
        int t = gidx >> 1, o = gidx & 1;
        float s = b_fc[o];
        for (int bb = 0; bb < NB; ++bb)
            s += partials[(size_t)(t * NB + bb) * 2 + o];
        out[gidx] = s;
    }
}

extern "C" void kernel_launch(void* const* d_in, const int* in_sizes, int n_in,
                              void* d_out, int out_size, void* d_ws, size_t ws_size,
                              hipStream_t stream) {
    (void)in_sizes; (void)n_in; (void)out_size; (void)ws_size;
    const float* features = (const float*)d_in[0];  // [1, 2048]
    const float* pc       = (const float*)d_in[1];  // [1, 1024, 2]
    const float* W_ih     = (const float*)d_in[2];  // [8192, 2]
    const float* W_hh     = (const float*)d_in[3];  // [8192, 2048]
    const float* b_ih     = (const float*)d_in[4];  // [8192]
    const float* b_hh     = (const float*)d_in[5];  // [8192]
    const float* W_fc     = (const float*)d_in[6];  // [2, 2048]
    const float* b_fc     = (const float*)d_in[7];  // [2]
    float* out = (float*)d_out;                     // [1, 1024, 2]

    char* ws = (char*)d_ws;
    __bf16*   Wswz     = (__bf16*)ws;
    __bf16*   hbuf     = (__bf16*)(ws + WSWZ_BYTES);
    unsigned* bar      = (unsigned*)(ws + BAR_OFF);
    float*    partials = (float*)(ws + PART_OFF);

    // 1) one-pass fp32 -> bf16 weight swizzle (16.7M elements)
    convert_whh_kernel<<<(G4 * H) / 256, 256, 0, stream>>>(W_hh, Wswz);
    // 2) h0 = features, reset barrier counter
    init_kernel<<<(H + 255) / 256, 256, 0, stream>>>(features, hbuf, bar);
    // 3) persistent LSTM decode
    lstm_persistent_kernel<<<NB, TPB, 0, stream>>>(
        Wswz, hbuf, pc, W_ih, b_ih, b_hh, W_fc, b_fc, partials, out, bar);
}
